// IsoMaxPlusSoftmax_63445256896531
// MI455X (gfx1250) — matrix operand order repaired and checked by reading, not compiled, not run on hardware
//
#include <hip/hip_runtime.h>
#include <hip/hip_bf16.h>

// ---------------------------------------------------------------------------
// IsoMax+ distances: out[n,c] = -|scale| * sqrt(max(2 - 2*<f_hat_n, p_hat_c>, 1e-12))
// N=16384, D=2048, C=8192.  Compute-bound GEMM -> bf16 split-accumulation
// (hi*hi + hi*lo + lo*hi) on v_wmma_f32_16x16x32_bf16, fp32 accumulate,
// normalization folded into the epilogue via precomputed 1/||row|| factors.
// ---------------------------------------------------------------------------

typedef __attribute__((ext_vector_type(8)))  __bf16 v8bf;
typedef __attribute__((ext_vector_type(16))) __bf16 v16bf;
typedef __attribute__((ext_vector_type(8)))  float  v8f;

#define N_ROWS 16384
#define D_DIM  2048
#define C_ROWS 8192

#define BM 128
#define BN 128
#define BK 32
#define LDS_STRIDE 40   // 32 bf16 + 8 pad -> 80B row stride, bank-conflict free, 16B aligned

// ---------------------------------------------------------------------------
// Kernel 1: one wave (32 lanes) per row -> reciprocal L2 norm into workspace.
// rn[0..N)   = 1/max(||features_row||, 1e-12)
// rn[N..N+C) = 1/max(||prototypes_row||, 1e-12)
// ---------------------------------------------------------------------------
__global__ void isomax_rnorm_kernel(const float* __restrict__ F,
                                    const float* __restrict__ P,
                                    float* __restrict__ rn) {
    const int wave = (int)((blockIdx.x * blockDim.x + threadIdx.x) >> 5);
    const int lane = (int)(threadIdx.x & 31);
    if (wave >= N_ROWS + C_ROWS) return;

    const float* src = (wave < N_ROWS)
                           ? (F + (size_t)wave * D_DIM)
                           : (P + (size_t)(wave - N_ROWS) * D_DIM);
    const float4* s4 = (const float4*)src;

    float ss = 0.0f;
#pragma unroll
    for (int i = 0; i < D_DIM / 4; i += 32) {     // 16 x float4 per lane
        float4 v = s4[i + lane];
        ss += v.x * v.x + v.y * v.y + v.z * v.z + v.w * v.w;
    }
#pragma unroll
    for (int off = 16; off > 0; off >>= 1)
        ss += __shfl_xor(ss, off, 32);

    if (lane == 0)
        rn[wave] = 1.0f / fmaxf(sqrtf(ss), 1e-12f);
}

// Split 16 fp32 values into bf16 hi/lo halves and store 2x16B each to LDS.
__device__ __forceinline__ void split_store16(float4 a0, float4 a1, float4 a2, float4 a3,
                                              __bf16* __restrict__ dhi,
                                              __bf16* __restrict__ dlo) {
    float x[16] = {a0.x, a0.y, a0.z, a0.w, a1.x, a1.y, a1.z, a1.w,
                   a2.x, a2.y, a2.z, a2.w, a3.x, a3.y, a3.z, a3.w};
    v8bf h0, h1, l0, l1;
#pragma unroll
    for (int j = 0; j < 8; ++j) {
        __bf16 h = (__bf16)x[j];
        h0[j] = h;
        l0[j] = (__bf16)(x[j] - (float)h);
    }
#pragma unroll
    for (int j = 0; j < 8; ++j) {
        __bf16 h = (__bf16)x[8 + j];
        h1[j] = h;
        l1[j] = (__bf16)(x[8 + j] - (float)h);
    }
    *(v8bf*)(dhi)     = h0;
    *(v8bf*)(dhi + 8) = h1;
    *(v8bf*)(dlo)     = l0;
    *(v8bf*)(dlo + 8) = l1;
}

union FragBF {
    v16bf v;
    v8bf  h[2];
};

// ---------------------------------------------------------------------------
// Kernel 2: 128x128 output tile per 256-thread block (8 wave32 waves, 2x4).
// Each wave: 64x32 sub-tile = 4x2 WMMA 16x16 tiles, f32 accumulators.
// Per K-step (BK=32): 3 split terms * 8 tiles = 24 v_wmma_f32_16x16x32_bf16.
// ---------------------------------------------------------------------------
__global__ __launch_bounds__(256)
void isomax_gemm_kernel(const float* __restrict__ F,
                        const float* __restrict__ P,
                        const float* __restrict__ scale_p,
                        const float* __restrict__ rnF,
                        const float* __restrict__ rnP,
                        float* __restrict__ out) {
    __shared__ __align__(16) __bf16 sAhi[BM * LDS_STRIDE];
    __shared__ __align__(16) __bf16 sAlo[BM * LDS_STRIDE];
    __shared__ __align__(16) __bf16 sBhi[BN * LDS_STRIDE];
    __shared__ __align__(16) __bf16 sBlo[BN * LDS_STRIDE];

    const int tid     = (int)threadIdx.x;
    const int wave    = tid >> 5;
    const int lane    = tid & 31;
    const int la      = lane & 15;     // position within 16-wide group
    const int hihalf  = lane >> 4;     // 0: lanes 0-15, 1: lanes 16-31

    const int wm = wave & 1;           // 2 wave rows  (64 rows each)
    const int wn = wave >> 1;          // 4 wave cols  (32 cols each)

    const int rowBase = (int)blockIdx.x * BM;
    const int colBase = (int)blockIdx.y * BN;

    // Cooperative load mapping: 256 threads x 16 floats = 128x32 block.
    const int lr = tid >> 1;            // 0..127: block-local row
    const int lk = (tid & 1) * 16;      // 0 or 16: k-offset

    const float* gA = F + (size_t)(rowBase + lr) * D_DIM + lk;
    const float* gB = P + (size_t)(colBase + lr) * D_DIM + lk;
    __bf16* wAhi = &sAhi[lr * LDS_STRIDE + lk];
    __bf16* wAlo = &sAlo[lr * LDS_STRIDE + lk];
    __bf16* wBhi = &sBhi[lr * LDS_STRIDE + lk];
    __bf16* wBlo = &sBlo[lr * LDS_STRIDE + lk];

    v8f acc[4][2] = {};

    for (int k0 = 0; k0 < D_DIM; k0 += BK) {
        // Global fp32 loads (b128), issued before the barrier for overlap.
        float4 a0 = *(const float4*)(gA + k0);
        float4 a1 = *(const float4*)(gA + k0 + 4);
        float4 a2 = *(const float4*)(gA + k0 + 8);
        float4 a3 = *(const float4*)(gA + k0 + 12);
        float4 b0 = *(const float4*)(gB + k0);
        float4 b1 = *(const float4*)(gB + k0 + 4);
        float4 b2 = *(const float4*)(gB + k0 + 8);
        float4 b3 = *(const float4*)(gB + k0 + 12);

        __syncthreads();   // previous iteration's fragment reads complete
        split_store16(a0, a1, a2, a3, wAhi, wAlo);
        split_store16(b0, b1, b2, b3, wBhi, wBlo);
        __syncthreads();   // LDS tile visible to all waves

        // A fragments (16x32, contiguous k order, same as B): lanes 0-15 hold K 0-15,
        // lanes 16-31 hold K 16-31, row M = lane&15.
        FragBF ahi[4], alo[4];
#pragma unroll
        for (int tm = 0; tm < 4; ++tm) {
            const int r = (wm * 64 + tm * 16 + la) * LDS_STRIDE + hihalf * 16;
            ahi[tm].h[0] = *(const v8bf*)&sAhi[r];
            ahi[tm].h[1] = *(const v8bf*)&sAhi[r + 8];
            alo[tm].h[0] = *(const v8bf*)&sAlo[r];
            alo[tm].h[1] = *(const v8bf*)&sAlo[r + 8];
        }
        // B fragments (32x16, contiguous k order, same as A): lanes 0-15 hold K 0-15,
        // lanes 16-31 hold K 16-31, column N = lane&15 (= prototype row).
        FragBF bhi[2], blo[2];
#pragma unroll
        for (int tn = 0; tn < 2; ++tn) {
            const int r = (wn * 32 + tn * 16 + la) * LDS_STRIDE + hihalf * 16;
            bhi[tn].h[0] = *(const v8bf*)&sBhi[r];
            bhi[tn].h[1] = *(const v8bf*)&sBhi[r + 8];
            blo[tn].h[0] = *(const v8bf*)&sBlo[r];
            blo[tn].h[1] = *(const v8bf*)&sBlo[r + 8];
        }

#pragma unroll
        for (int tm = 0; tm < 4; ++tm) {
#pragma unroll
            for (int tn = 0; tn < 2; ++tn) {
                acc[tm][tn] = __builtin_amdgcn_wmma_f32_16x16x32_bf16(
                    false, ahi[tm].v, false, bhi[tn].v, (short)0, acc[tm][tn], false, false);
                acc[tm][tn] = __builtin_amdgcn_wmma_f32_16x16x32_bf16(
                    false, ahi[tm].v, false, blo[tn].v, (short)0, acc[tm][tn], false, false);
                acc[tm][tn] = __builtin_amdgcn_wmma_f32_16x16x32_bf16(
                    false, alo[tm].v, false, bhi[tn].v, (short)0, acc[tm][tn], false, false);
            }
        }
    }

    // Epilogue: C/D 16x16 f32 layout -> VGPR v, lane half: M = v + 8*half, N = lane&15.
    const float scale = fabsf(scale_p[0]);
#pragma unroll
    for (int tm = 0; tm < 4; ++tm) {
#pragma unroll
        for (int tn = 0; tn < 2; ++tn) {
            const int c  = colBase + wn * 32 + tn * 16 + la;
            const float rp = rnP[c];
#pragma unroll
            for (int v = 0; v < 8; ++v) {
                const int m = rowBase + wm * 64 + tm * 16 + hihalf * 8 + v;
                const float s  = acc[tm][tn][v] * rnF[m] * rp;
                const float sq = fmaxf(2.0f - 2.0f * s, 1e-12f);
                out[(size_t)m * C_ROWS + c] = -scale * sqrtf(sq);
            }
        }
    }
}

// ---------------------------------------------------------------------------
extern "C" void kernel_launch(void* const* d_in, const int* in_sizes, int n_in,
                              void* d_out, int out_size, void* d_ws, size_t ws_size,
                              hipStream_t stream) {
    const float* F = (const float*)d_in[0];   // [N, D] fp32
    const float* P = (const float*)d_in[1];   // [C, D] fp32
    const float* S = (const float*)d_in[2];   // [1]    fp32
    float* out = (float*)d_out;               // [N, C] fp32

    float* rn  = (float*)d_ws;                // N + C reciprocal norms (96 KB)
    float* rnF = rn;
    float* rnP = rn + N_ROWS;

    // 1 wave per row, 8 waves per block.
    const int totalRows = N_ROWS + C_ROWS;
    isomax_rnorm_kernel<<<dim3(totalRows / 8), dim3(256), 0, stream>>>(F, P, rn);

    dim3 grid(N_ROWS / BM, C_ROWS / BN);      // 128 x 64
    isomax_gemm_kernel<<<grid, dim3(256), 0, stream>>>(F, P, S, rnF, rnP, out);
}
